// AxialAttention_2740189135099
// MI455X (gfx1250) — matrix operand order repaired and checked by reading, not compiled, not run on hardware
//
#include <hip/hip_runtime.h>

// ---------------------------------------------------------------------------
// AxialAttention pipeline for gfx1250 (MI455X).
// GEMMs via v_wmma_f32_16x16x32_bf16. Data movement:
//   - TDM (tensor_load_to_lds + s_wait_tensorcnt) for the attn q-slab
//   - global_load_async_to_lds_b128 + s_wait_asynccnt for the P slab
//   - vectorized b128 + packed LDS stores for fp32->bf16 conv staging
// Fragment layouts per CDNA5 ISA 7.12.2; D# per ISA 8.3/8.4.
// ---------------------------------------------------------------------------

typedef __bf16 bf16;
typedef __attribute__((ext_vector_type(16))) __bf16 v16bf;
typedef __attribute__((ext_vector_type(8)))  __bf16 bf16x8;
typedef __attribute__((ext_vector_type(4)))  __bf16 bf16x4;
typedef __attribute__((ext_vector_type(8)))  float  v8f;
typedef __attribute__((ext_vector_type(4)))  float  f32x4;
typedef __attribute__((ext_vector_type(2)))  float  f32x2;
typedef __attribute__((ext_vector_type(4)))  unsigned int u32x4;
typedef __attribute__((ext_vector_type(8)))  int    i32x8;
typedef __attribute__((ext_vector_type(4)))  int    i32x4;

#define NB   16
#define CCH  32
#define HWP  65536          // 256*256
#define CHW  (CCH * HWP)    // 2097152 elements per batch

#if defined(__has_builtin)
#if __has_builtin(__builtin_amdgcn_tensor_load_to_lds)
#define HAVE_TDM 1
#endif
#if __has_builtin(__builtin_amdgcn_s_cluster_barrier)
#define HAVE_CLB 1
#endif
#endif

// A-matrix (16x32 bf16) per-lane K offset for vgpr-half j (0..7):
__device__ __forceinline__ int a_koff(int j, int hi) {
    return 2 * j + (j >= 4 ? 8 : 0) + (hi ? 8 : 0);
}

// Async DMA: global (16B) -> LDS. VDST operand = LDS byte offset; gfx1250 LDS
// mapping truncates a flat LDS address to addr[31:0] (ISA 10.2).
__device__ __forceinline__ void async_copy16(const bf16* g, bf16* l) {
    asm volatile("global_load_async_to_lds_b128 %0, %1, off"
                 :: "v"((unsigned)(uintptr_t)l), "v"(g)
                 : "memory");
}
#define WAIT_ASYNC_0() asm volatile("s_wait_asynccnt 0x0" ::: "memory")

#ifdef HAVE_TDM
// TDM 2D tile load: bf16 tile (tile_d0 x tile_d1) from row-major tensor
// (stride0 elements per row) into LDS at byte offset ldsOff. D# per ISA 8.3/8.4.
// This toolchain exposes the 6-arg builtin: (g0, g1, g2, g3, g4, cpol).
__device__ __forceinline__ void tdm_load_2d(const bf16* gtile, unsigned ldsOff,
                                            int tensor_d0, int tensor_d1,
                                            int tile_d0, int tile_d1, int stride0)
{
    const unsigned long long ga = (unsigned long long)(uintptr_t)gtile;
    u32x4 g0;
    g0[0] = 1u;                                                   // count=1 (valid), user
    g0[1] = ldsOff;                                               // lds_addr (bytes)
    g0[2] = (unsigned)(ga & 0xFFFFFFFFull);                       // global_addr[31:0]
    g0[3] = (unsigned)((ga >> 32) & 0x01FFFFFFull) | (2u << 30);  // [56:32] | type=2
    i32x8 g1;
    g1[0] = (int)(1u << 16);                                      // data_size=1 (2B)
    g1[1] = (int)((unsigned)tensor_d0 << 16);                     // tensor_dim0 lo16
    g1[2] = (int)(((unsigned)tensor_d0 >> 16) | ((unsigned)tensor_d1 << 16));
    g1[3] = (int)(((unsigned)tensor_d1 >> 16) | ((unsigned)tile_d0 << 16));
    g1[4] = (int)(unsigned)tile_d1;                               // tile_dim1, tile_dim2=0
    g1[5] = (int)(unsigned)stride0;                               // tensor_dim0_stride
    g1[6] = 0;
    g1[7] = 0;
    i32x4 gz4 = {};
    i32x8 gz8 = {};
    __builtin_amdgcn_tensor_load_to_lds(g0, g1, gz4, gz4, gz8, 0);
}
#endif

// Build A fragment from a row-major bf16 row pointer: K-runs are contiguous.
__device__ __forceinline__ v16bf load_a_frag(const bf16* rowp, int kbase, int hi) {
    bf16x8 alo = *(const bf16x8*)(rowp + kbase + hi * 8);
    bf16x8 ahi = *(const bf16x8*)(rowp + kbase + 16 + hi * 8);
    v16bf A;
#pragma unroll
    for (int e = 0; e < 8; ++e) { A[e] = alo[e]; A[8 + e] = ahi[e]; }
    return A;
}

// ---------------------------------------------------------------------------
// conv 1x1 (32->32) + BN + ReLU, fp32 in -> bf16 out.
// grid = NB * 1024 blocks of 256
// ---------------------------------------------------------------------------
__global__ void __launch_bounds__(256)
k_conv1x1(const float* __restrict__ in, const float* __restrict__ wgt,
          const float* __restrict__ sc, const float* __restrict__ bi,
          bf16* __restrict__ out)
{
    __shared__ bf16 slab[32 * 64];

    const int tid  = threadIdx.x;
    const int lane = tid & 31;
    const int w    = tid >> 5;
    const int pb   = blockIdx.x & 1023;
    const int b    = blockIdx.x >> 10;
    const int p0b  = pb * 64;

    const float* inb = in + (size_t)b * CHW;

#pragma unroll
    for (int it = 0; it < 2; ++it) {
        const int c    = tid + it * 256;
        const int row  = c >> 4;
        const int chk  = c & 15;
        f32x4 d = *(const f32x4*)(inb + (size_t)row * HWP + p0b + chk * 4);
        bf16x4 h;
        h[0] = (bf16)d.x; h[1] = (bf16)d.y; h[2] = (bf16)d.z; h[3] = (bf16)d.w;
        *(bf16x4*)(&slab[row * 64 + chk * 4]) = h;
    }
    __syncthreads();

    const int m  = lane & 15;
    const int hi = lane >> 4;
    const int n  = lane & 15;
    const int rb = w & 1;
    const int px0 = (w >> 1) * 16;
    const int row = rb * 16 + m;
    const int kb2 = hi ? 16 : 0;

    v16bf A;
#pragma unroll
    for (int j = 0; j < 8; ++j) {
        const int k0 = a_koff(j, hi);
        f32x2 wp = *(const f32x2*)(wgt + row * CCH + k0);
        A[2 * j]     = (bf16)wp.x;
        A[2 * j + 1] = (bf16)wp.y;
    }
    v16bf Bm;
#pragma unroll
    for (int j = 0; j < 8; ++j) {
        const int k0 = a_koff(j, hi);
        Bm[2 * j]     = slab[k0 * 64 + px0 + n];
        Bm[2 * j + 1] = slab[(k0 + 1) * 64 + px0 + n];
    }
    v8f acc = {};
    acc = __builtin_amdgcn_wmma_f32_16x16x32_bf16(false, A, false, Bm,
                                                  (short)0, acc, false, false);
#pragma unroll
    for (int r = 0; r < 8; ++r) {
        const int mm = r + (hi ? 8 : 0);
        const int co = rb * 16 + mm;
        float y = acc[r] * sc[co] + bi[co];
        y = y > 0.f ? y : 0.f;
        out[(size_t)b * CHW + (size_t)co * HWP + p0b + px0 + n] = (bf16)y;
    }
}

// ---------------------------------------------------------------------------
// conv 3x3 pad=1 + BN + ReLU, bf16 in -> fp32 out. Implicit GEMM, K = 9*32.
// Block = 64 px of one image row; stage 32ch x 3row x 96col halo slab to LDS
// (b128 loads; chunk-granular OOB zeroing), 8 waves = 2 rb x 4 px tiles.
// grid = NB * 1024 blocks of 256
// ---------------------------------------------------------------------------
__global__ void __launch_bounds__(256)
k_conv3x3(const bf16* __restrict__ in, const float* __restrict__ wgt,
          const float* __restrict__ sc, const float* __restrict__ bi,
          float* __restrict__ out)
{
    __shared__ bf16 slab[32 * 3 * 96];   // [ch][dy][col], col 0 == w0c-16

    const int tid  = threadIdx.x;
    const int lane = tid & 31;
    const int w    = tid >> 5;
    const int pb   = blockIdx.x & 1023;
    const int b    = blockIdx.x >> 10;
    const int p0b  = pb * 64;
    const int h    = p0b >> 8;
    const int w0c  = p0b & 255;          // 0,64,128,192

    const bf16* inb = in + (size_t)b * CHW;
    const bf16x8 zed = {};

    // stage: 1152 chunks of 8 bf16 (32ch x 3rows x 12 chunks of 8 cols)
    for (int it = 0; it < 5; ++it) {
        const int c = tid + it * 256;
        if (c < 1152) {
            const int ch  = c / 36;
            const int rem = c % 36;
            const int dy  = rem / 12;
            const int ck  = rem % 12;
            const int hh  = h + dy - 1;
            const int gc0 = w0c - 16 + ck * 8;        // multiple of 8 -> all-or-nothing
            const bool ok = (hh >= 0) && (hh < 256) && (gc0 >= 0) && (gc0 < 256);
            const int hcl = hh < 0 ? 0 : (hh > 255 ? 255 : hh);
            const int ccl = gc0 < 0 ? 0 : (gc0 > 248 ? 248 : gc0);
            bf16x8 d = *(const bf16x8*)(inb + (size_t)ch * HWP + hcl * 256 + ccl);
            if (!ok) d = zed;
            *(bf16x8*)(&slab[(ch * 3 + dy) * 96 + ck * 8]) = d;
        }
    }
    __syncthreads();

    const int m  = lane & 15;
    const int hi = lane >> 4;
    const int n  = lane & 15;
    const int rb = w & 1;
    const int pt = w >> 1;               // 0..3
    const int row = rb * 16 + m;
    const int kb2 = hi ? 16 : 0;
    const int colb = 16 + pt * 16 + n;   // slab col for dx=1

    v8f acc = {};
#pragma unroll
    for (int dy = 0; dy < 3; ++dy) {
#pragma unroll
        for (int dx = 0; dx < 3; ++dx) {
            v16bf A;
#pragma unroll
            for (int j = 0; j < 8; ++j) {
                const int k0 = a_koff(j, hi);
                A[2 * j]     = (bf16)wgt[(row * CCH + k0) * 9 + dy * 3 + dx];
                A[2 * j + 1] = (bf16)wgt[(row * CCH + k0 + 1) * 9 + dy * 3 + dx];
            }
            const int col = colb + dx - 1;
            v16bf Bm;
#pragma unroll
            for (int j = 0; j < 8; ++j) {
                const int k0 = a_koff(j, hi);
                Bm[2 * j]     = slab[(k0 * 3 + dy) * 96 + col];
                Bm[2 * j + 1] = slab[((k0 + 1) * 3 + dy) * 96 + col];
            }
            acc = __builtin_amdgcn_wmma_f32_16x16x32_bf16(false, A, false, Bm,
                                                          (short)0, acc, false, false);
        }
    }
#pragma unroll
    for (int r = 0; r < 8; ++r) {
        const int mm = r + (hi ? 8 : 0);
        const int co = rb * 16 + mm;
        float y = acc[r] * sc[co] + bi[co];
        y = y > 0.f ? y : 0.f;
        out[(size_t)b * CHW + (size_t)co * HWP + p0b + pt * 16 + n] = y;
    }
}

// ---------------------------------------------------------------------------
// Fused q & k 1x1 convs (4+4 ch in one 8-row WMMA tile), fp32 in.
// q stored (B,1024,256) bf16; k stored transposed kT[b][w][c8*256+h].
// grid = NB * 512 blocks of 256
// ---------------------------------------------------------------------------
__global__ void __launch_bounds__(256)
k_qk(const float* __restrict__ in,
     const float* __restrict__ qw, const float* __restrict__ qs, const float* __restrict__ qbb,
     const float* __restrict__ kw, const float* __restrict__ ks, const float* __restrict__ kbb,
     bf16* __restrict__ q, bf16* __restrict__ kT)
{
    __shared__ bf16 slab[32 * 128];

    const int tid  = threadIdx.x;
    const int lane = tid & 31;
    const int w    = tid >> 5;
    const int pb   = blockIdx.x & 511;
    const int b    = blockIdx.x >> 9;
    const int p0b  = pb * 128;

    const float* inb = in + (size_t)b * CHW;

#pragma unroll
    for (int it = 0; it < 4; ++it) {
        const int c   = tid + it * 256;
        const int row = c >> 5;
        const int chk = c & 31;
        f32x4 d = *(const f32x4*)(inb + (size_t)row * HWP + p0b + chk * 4);
        bf16x4 hh;
        hh[0] = (bf16)d.x; hh[1] = (bf16)d.y; hh[2] = (bf16)d.z; hh[3] = (bf16)d.w;
        *(bf16x4*)(&slab[row * 128 + chk * 4]) = hh;
    }
    __syncthreads();

    const int m  = lane & 15;
    const int hi = lane >> 4;
    const int n  = lane & 15;
    const int px0 = w * 16;
    const int kb2 = hi ? 16 : 0;

    const float* wrow = (m < 4) ? (qw + m * CCH) : (kw + ((m - 4) & 3) * CCH);
    v16bf A;
#pragma unroll
    for (int j = 0; j < 8; ++j) {
        const int k0 = a_koff(j, hi);
        f32x2 wp = *(const f32x2*)(wrow + k0);
        A[2 * j]     = (bf16)((m < 8) ? wp.x : 0.f);
        A[2 * j + 1] = (bf16)((m < 8) ? wp.y : 0.f);
    }
    v16bf Bm;
#pragma unroll
    for (int j = 0; j < 8; ++j) {
        const int k0 = a_koff(j, hi);
        Bm[2 * j]     = slab[k0 * 128 + px0 + n];
        Bm[2 * j + 1] = slab[(k0 + 1) * 128 + px0 + n];
    }
    v8f acc = {};
    acc = __builtin_amdgcn_wmma_f32_16x16x32_bf16(false, A, false, Bm,
                                                  (short)0, acc, false, false);
    const int p  = p0b + px0 + n;
    const int hc = p >> 8;
    const int wc = p & 255;
#pragma unroll
    for (int r = 0; r < 8; ++r) {
        const int mm = r + (hi ? 8 : 0);
        if (mm < 8) {
            const bool isq = mm < 4;
            const int  c8  = isq ? mm : (mm - 4);
            const float s  = isq ? qs[c8] : ks[c8];
            const float bb = isq ? qbb[c8] : kbb[c8];
            float y = acc[r] * s + bb;
            y = y > 0.f ? y : 0.f;
            if (isq) q [(size_t)b * 262144 + (size_t)c8 * 65536 + p] = (bf16)y;
            else     kT[(size_t)b * 262144 + (size_t)wc * 1024 + c8 * 256 + hc] = (bf16)y;
        }
    }
}

// ---------------------------------------------------------------------------
// P = sigmoid(kT @ q): (256x1024)x(1024x256) -> bf16 (256x256) per batch.
// Block: 16(d) x 128(a). The shared q slab (32K x 128N bf16 per step) is
// DMA'd by the Tensor Data Mover (wave 0 issues, double-buffered,
// s_wait_tensorcnt + barriers). grid = NB * 32 blocks of 256
// ---------------------------------------------------------------------------
__global__ void __launch_bounds__(256)
k_attn(const bf16* __restrict__ kT, const bf16* __restrict__ q, bf16* __restrict__ P)
{
    __shared__ bf16 sq[2][32 * 128];

    const int tid  = threadIdx.x;
    const int lane = tid & 31;
    const int w    = tid >> 5;
    const int ab   = blockIdx.x & 1;
    const int dt   = (blockIdx.x >> 1) & 15;
    const int b    = blockIdx.x >> 5;
    const int d0   = dt * 16;
    const int aB   = ab * 128;

    const bf16* kTb = kT + (size_t)b * 262144;
    const bf16* qb  = q  + (size_t)b * 262144;

    const int m  = lane & 15;
    const int hi = lane >> 4;
    const int n  = lane & 15;
    const int kb2 = hi ? 16 : 0;
    const bf16* arow = kTb + (size_t)(d0 + m) * 1024;

#ifdef HAVE_TDM
    const unsigned lds0 = (unsigned)(uintptr_t)&sq[0][0];
    const unsigned lds1 = (unsigned)(uintptr_t)&sq[1][0];
    if (w == 0) {
        // prologue: slab for kk=0 (tile 128 x 32 of q, rows 0..31, cols aB..aB+127)
        tdm_load_2d(qb + aB, lds0, 256, 1024, 128, 32, 256);
    }
#else
    const int srow = tid >> 3;
    const int scp  = (tid & 7) * 2;
    {
        const bf16* g0 = qb + (size_t)srow * 256 + aB + scp * 8;
        bf16* l0 = &sq[0][srow * 128 + scp * 8];
        async_copy16(g0, l0);
        async_copy16(g0 + 8, l0 + 8);
    }
#endif

    v8f acc = {};
    for (int kk = 0; kk < 32; ++kk) {
        const int cur = kk & 1;
#ifdef HAVE_TDM
        if (w == 0) {
            if (kk + 1 < 32) {
                tdm_load_2d(qb + (size_t)(kk + 1) * 8192 + aB,
                            cur ? lds0 : lds1, 256, 1024, 128, 32, 256);
                __builtin_amdgcn_s_wait_tensorcnt(1);
            } else {
                __builtin_amdgcn_s_wait_tensorcnt(0);
            }
        }
#else
        if (kk + 1 < 32) {
            const bf16* g0 = qb + (size_t)(kk + 1) * 8192 + (size_t)srow * 256 + aB + scp * 8;
            bf16* l0 = &sq[cur ^ 1][srow * 128 + scp * 8];
            async_copy16(g0, l0);
            async_copy16(g0 + 8, l0 + 8);
            asm volatile("s_wait_asynccnt 0x2" ::: "memory");
        } else {
            WAIT_ASYNC_0();
        }
#endif
        __syncthreads();

        v16bf A = load_a_frag(arow, kk * 32, hi);
        v16bf Bm;
#pragma unroll
        for (int j = 0; j < 8; ++j) {
            const int k0 = a_koff(j, hi);
            Bm[2 * j]     = sq[cur][k0 * 128 + w * 16 + n];
            Bm[2 * j + 1] = sq[cur][(k0 + 1) * 128 + w * 16 + n];
        }
        acc = __builtin_amdgcn_wmma_f32_16x16x32_bf16(false, A, false, Bm,
                                                      (short)0, acc, false, false);
        __syncthreads();
    }

#ifdef HAVE_CLB
    __builtin_amdgcn_s_cluster_barrier();   // NOP when not in a cluster (ISA 6.6)
#endif

    bf16* Pb = P + (size_t)b * 65536;
    const int a0 = aB + w * 16;
#pragma unroll
    for (int r = 0; r < 8; ++r) {
        const int mm = r + (hi ? 8 : 0);
        const float sgm = 1.0f / (1.0f + __expf(-acc[r]));
        Pb[(d0 + mm) * 256 + a0 + n] = (bf16)sgm;
    }
}

// ---------------------------------------------------------------------------
// out = gamma * (v @ P) + res : (8192x256)x(256x256) per batch, fp32 out.
// Block: 128(m) x 16(a); full K=256 P column-block (8KB) staged once via
// global_load_async_to_lds_b128. grid = NB*64*16 = 16384
// ---------------------------------------------------------------------------
__global__ void __launch_bounds__(256)
k_av(const bf16* __restrict__ v, const bf16* __restrict__ P,
     const float* __restrict__ res, const float* __restrict__ gamma,
     float* __restrict__ out)
{
    __shared__ bf16 sP[256 * 16];

    const int tid  = threadIdx.x;
    const int lane = tid & 31;
    const int w    = tid >> 5;
    const int at   = blockIdx.x & 15;
    const int mb   = (blockIdx.x >> 4) & 63;
    const int b    = blockIdx.x >> 10;
    const int a0   = at * 16;
    const int m0   = mb * 128 + w * 16;

    const bf16* vb = v + (size_t)b * CHW;
    const bf16* Pb = P + (size_t)b * 65536;

    {
        const bf16* g0 = Pb + (size_t)tid * 256 + a0;
        bf16* l0 = &sP[tid * 16];
        async_copy16(g0, l0);
        async_copy16(g0 + 8, l0 + 8);
    }
    WAIT_ASYNC_0();
    __syncthreads();

    const int m  = lane & 15;
    const int hi = lane >> 4;
    const int n  = lane & 15;
    const int kb2 = hi ? 16 : 0;
    const bf16* arow = vb + (size_t)(m0 + m) * 256;

    v8f acc = {};
#pragma unroll
    for (int kk = 0; kk < 8; ++kk) {
        const int kbase = kk * 32;
        v16bf A = load_a_frag(arow, kbase, hi);
        v16bf Bm;
#pragma unroll
        for (int j = 0; j < 8; ++j) {
            const int k0 = kbase + a_koff(j, hi);
            Bm[2 * j]     = sP[k0 * 16 + n];
            Bm[2 * j + 1] = sP[(k0 + 1) * 16 + n];
        }
        acc = __builtin_amdgcn_wmma_f32_16x16x32_bf16(false, A, false, Bm,
                                                      (short)0, acc, false, false);
    }
    const float g = gamma[0];
    const size_t base = (size_t)b * CHW;
#pragma unroll
    for (int r = 0; r < 8; ++r) {
        const int mm = r + (hi ? 8 : 0);
        const size_t idx = base + (size_t)(m0 + mm) * 256 + a0 + n;
        out[idx] = g * acc[r] + res[idx];
    }
}

// ---------------------------------------------------------------------------
extern "C" void kernel_launch(void* const* d_in, const int* in_sizes, int n_in,
                              void* d_out, int out_size, void* d_ws, size_t ws_size,
                              hipStream_t stream)
{
    (void)in_sizes; (void)n_in; (void)out_size; (void)ws_size;

    const float* x     = (const float*)d_in[0];
    const float* c0_w  = (const float*)d_in[1];
    const float* c0_s  = (const float*)d_in[2];
    const float* c0_b  = (const float*)d_in[3];
    const float* c1_w  = (const float*)d_in[4];
    const float* c1_s  = (const float*)d_in[5];
    const float* c1_b  = (const float*)d_in[6];
    const float* hq_w  = (const float*)d_in[7];
    const float* hq_s  = (const float*)d_in[8];
    const float* hq_b  = (const float*)d_in[9];
    const float* hk_w  = (const float*)d_in[10];
    const float* hk_s  = (const float*)d_in[11];
    const float* hk_b  = (const float*)d_in[12];
    const float* hv_w  = (const float*)d_in[13];
    const float* hv_s  = (const float*)d_in[14];
    const float* hv_b  = (const float*)d_in[15];
    const float* gam_h = (const float*)d_in[16];
    const float* wq_w  = (const float*)d_in[17];
    const float* wq_s  = (const float*)d_in[18];
    const float* wq_b  = (const float*)d_in[19];
    const float* wk_w  = (const float*)d_in[20];
    const float* wk_s  = (const float*)d_in[21];
    const float* wk_b  = (const float*)d_in[22];
    const float* wv_w  = (const float*)d_in[23];
    const float* wv_s  = (const float*)d_in[24];
    const float* wv_b  = (const float*)d_in[25];
    const float* gam_w = (const float*)d_in[26];
    float* out = (float*)d_out;

    char* ws = (char*)d_ws;
    float* y1  = (float*)(ws);
    float* hx  = (float*)(ws + 134217728);
    bf16*  vb  = (bf16*) (ws + 268435456);
    bf16*  qB  = (bf16*) (ws + 335544320);
    bf16*  kTB = (bf16*) (ws + 343932928);
    bf16*  PB  = (bf16*) (ws + 352321536);

    const dim3 blk(256);

    k_conv1x1<<<16384, blk, 0, stream>>>(x, c0_w, c0_s, c0_b, vb);
    k_conv3x3<<<16384, blk, 0, stream>>>(vb, c1_w, c1_s, c1_b, y1);

    // ---- height-axis attention (input y1) ----
    k_qk  <<<8192,  blk, 0, stream>>>(y1, hq_w, hq_s, hq_b, hk_w, hk_s, hk_b, qB, kTB);
    k_conv1x1<<<16384, blk, 0, stream>>>(y1, hv_w, hv_s, hv_b, vb);
    k_attn<<<512,   blk, 0, stream>>>(kTB, qB, PB);
    k_av  <<<16384, blk, 0, stream>>>(vb, PB, y1, gam_h, hx);

    // ---- width-axis attention (input hx) ----
    k_qk  <<<8192,  blk, 0, stream>>>(hx, wq_w, wq_s, wq_b, wk_w, wk_s, wk_b, qB, kTB);
    k_conv1x1<<<16384, blk, 0, stream>>>(hx, wv_w, wv_s, wv_b, vb);
    k_attn<<<512,   blk, 0, stream>>>(kTB, qB, PB);
    k_av  <<<16384, blk, 0, stream>>>(vb, PB, hx, gam_w, out);
}
